// HiRNNDecoder_68298569940993
// MI455X (gfx1250) — matrix operand order repaired and checked by reading, not compiled, not run on hardware
//
#include <hip/hip_runtime.h>
#include <hip/hip_bf16.h>

// ---------------------------------------------------------------------------
// HiRNN decoder step for MI455X (gfx1250, wave32, WMMA).
// Sizes (hardcoded from reference): B=64 S=400 H=1024 M=1024 E=512 V=32000 A=8
// ---------------------------------------------------------------------------

#define Bsz 64
#define Ssz 400
#define Hsz 1024
#define Msz 1024
#define Esz 512
#define Vsz 32000
#define Asz 8
#define G3H 3072          // 3*H
#define HM  2048          // H+M

typedef __attribute__((ext_vector_type(16))) _Float16 v16h;
typedef __attribute__((ext_vector_type(4)))  _Float16 v4h;
typedef __attribute__((ext_vector_type(8)))  float    v8f;

// ---------------------------------------------------------------------------
// WMMA GEMM:  C[rows, N] = A[rows, K] * W[N, K]^T + bias[N]
//  * 128 threads = 4 waves; each wave owns a 16x32 output slice -> 128 cols
//    per block (2 WMMAs per wave per 32-k step, A fragment reused).
//  * A tile (16x32) staged to LDS as f16, double-buffered, one barrier/step.
//  * Row gather (aspect routing) hoisted out of the K loop; rowList is always
//    valid (identity list for non-routed calls).
//  * B rows streamed from HBM as f32, cvt to f16 in regs; next tile prefetched.
// ---------------------------------------------------------------------------
__global__ void __launch_bounds__(128)
gemm_wmma_f16(const float* __restrict__ A, int lda,
              const float* __restrict__ W,
              const float* __restrict__ bias,
              float* __restrict__ C, int ldc,
              int Mrows, int N, int K,
              const int* __restrict__ rowList,   // identity for non-routed
              const int* __restrict__ counts,    // null => non-routed
              long long wStride, long long bStride)
{
    __shared__ __attribute__((aligned(16))) _Float16 As[2][16][36];

    const int wave = threadIdx.x >> 5;
    const int lane = threadIdx.x & 31;

    int cnt = Mrows;
    const int*   rl = rowList;
    const float* Wp = W;
    const float* bp = bias;
    if (counts) {                                 // aspect-routed mode
        const int a = blockIdx.z;
        cnt = counts[a];
        rl  = rowList + a * Bsz;
        Wp  = W + (size_t)a * (size_t)wStride;
        if (bias) bp = bias + (size_t)a * (size_t)bStride;
    }
    const int m0 = blockIdx.y * 16;
    if (m0 >= cnt) return;                        // uniform across block

    const int n0      = blockIdx.x * 128 + wave * 32;
    const int halfsel = lane >> 4;                // 0: lanes 0-15, 1: lanes 16-31
    const int l16     = lane & 15;
    const int ncol0   = n0 + l16;
    const int ncol1   = n0 + 16 + l16;

    // --- hoisted A staging assignment: thread -> (row, float4 column) ---
    const int ar = threadIdx.x >> 3;              // 0..15
    const int ak = (threadIdx.x & 7) << 2;        // 0,4,...,28
    const float* arow = nullptr;
    {
        const int agm = m0 + ar;
        if (agm < cnt) arow = A + (size_t)rl[agm] * lda + ak;
    }

    // --- hoisted B row pointers (this lane's K phase) ---
    const float* wrow0 = Wp + (size_t)ncol0 * K + halfsel * 16;
    const float* wrow1 = Wp + (size_t)ncol1 * K + halfsel * 16;

    auto stageA = [&](int kb, int buf) {
        float4 v = make_float4(0.f, 0.f, 0.f, 0.f);
        if (arow) v = *(const float4*)(arow + kb);
        v4h hv;
        hv[0] = (_Float16)v.x; hv[1] = (_Float16)v.y;
        hv[2] = (_Float16)v.z; hv[3] = (_Float16)v.w;
        *(v4h*)(&As[buf][ar][ak]) = hv;
    };

    auto loadB = [&](const float* wr, int kb) -> v16h {
        const float4* w4 = (const float4*)(wr + kb);
        const float4 w0 = w4[0], w1 = w4[1], w2 = w4[2], w3 = w4[3];
        v16h bf;
        bf[0]  = (_Float16)w0.x; bf[1]  = (_Float16)w0.y;
        bf[2]  = (_Float16)w0.z; bf[3]  = (_Float16)w0.w;
        bf[4]  = (_Float16)w1.x; bf[5]  = (_Float16)w1.y;
        bf[6]  = (_Float16)w1.z; bf[7]  = (_Float16)w1.w;
        bf[8]  = (_Float16)w2.x; bf[9]  = (_Float16)w2.y;
        bf[10] = (_Float16)w2.z; bf[11] = (_Float16)w2.w;
        bf[12] = (_Float16)w3.x; bf[13] = (_Float16)w3.y;
        bf[14] = (_Float16)w3.z; bf[15] = (_Float16)w3.w;
        return bf;
    };

    v8f acc0 = {}, acc1 = {};
    const int nK = K >> 5;                        // K is a multiple of 32

    stageA(0, 0);
    __syncthreads();

    int p = 0;
    for (int kt = 0; kt < nK; ++kt) {
        const int kb = kt << 5;

        // A fragment from LDS (contiguous k order: k = 16*half + s, matches B)
        v16h af;
        {
            const int m    = l16;
            const int koff = halfsel * 16;
            #pragma unroll
            for (int j = 0; j < 8; ++j) af[j]     = As[p][m][koff + j];
            #pragma unroll
            for (int j = 0; j < 8; ++j) af[8 + j] = As[p][m][koff + 8 + j];
        }

        // B fragments (two 16-col tiles share the A fragment)
        const v16h bf0 = loadB(wrow0, kb);
        const v16h bf1 = loadB(wrow1, kb);

        if (kt + 1 < nK) {
            stageA(kb + 32, 1 - p);
            __builtin_prefetch(wrow0 + kb + 32, 0, 1);   // global_prefetch_b8
            __builtin_prefetch(wrow1 + kb + 32, 0, 1);
        }
        __syncthreads();

        acc0 = __builtin_amdgcn_wmma_f32_16x16x32_f16(
                   false, af, false, bf0, (short)0, acc0, false, false);
        acc1 = __builtin_amdgcn_wmma_f32_16x16x32_f16(
                   false, af, false, bf1, (short)0, acc1, false, false);
        p = 1 - p;
    }

    // --- store C (rows 0-7 in lanes 0-15, rows 8-15 in lanes 16-31) ---
    const float bv0 = bp ? bp[ncol0] : 0.f;
    const float bv1 = bp ? bp[ncol1] : 0.f;
    #pragma unroll
    for (int g = 0; g < 8; ++g) {
        const int gm = m0 + halfsel * 8 + g;
        if (gm < cnt) {
            const size_t grow = (size_t)rl[gm];
            C[grow * ldc + ncol0] = acc0[g] + bv0;
            C[grow * ldc + ncol1] = acc1[g] + bv1;
        }
    }
}

// ---------------------------------------------------------------------------
// Aspect routing table + identity list.
// ---------------------------------------------------------------------------
__global__ void aspect_build(const int* __restrict__ aspect_id,
                             int* __restrict__ counts, int* __restrict__ lists,
                             int* __restrict__ idlist)
{
    __shared__ int cnt[Asz];
    const int t = threadIdx.x;
    if (t < Asz) cnt[t] = 0;
    __syncthreads();
    int a = 0, pos = 0;
    if (t < Bsz) { a = aspect_id[t]; pos = atomicAdd(&cnt[a], 1); }
    __syncthreads();
    if (t < Bsz) { lists[a * Bsz + pos] = t; idlist[t] = t; }
    if (t < Asz) counts[t] = cnt[t];
}

// ---------------------------------------------------------------------------
// GRU pointwise stage 1: t_h = GRUCell(h, h_t_init); h_sel = peos ? t_h : h
// ---------------------------------------------------------------------------
__global__ void __launch_bounds__(256)
gru1_select(const float* __restrict__ gi, const float* __restrict__ gh,
            const float* __restrict__ h,  const float* __restrict__ h_t_init,
            const int* __restrict__ y,    const float* __restrict__ sent_in,
            float* __restrict__ h_sel,    float* __restrict__ sent_out)
{
    const int idx = blockIdx.x * blockDim.x + threadIdx.x;
    if (idx >= Bsz * Hsz) return;
    const int b = idx >> 10, j = idx & (Hsz - 1);
    const float ir = gi[b * G3H + j],            hr = gh[b * G3H + j];
    const float iz = gi[b * G3H + Hsz + j],      hz = gh[b * G3H + Hsz + j];
    const float in_ = gi[b * G3H + 2 * Hsz + j], hn = gh[b * G3H + 2 * Hsz + j];
    const float r = 1.f / (1.f + __expf(-(ir + hr)));
    const float z = 1.f / (1.f + __expf(-(iz + hz)));
    const float n = tanhf(in_ + r * hn);
    const float th = (1.f - z) * n + z * h_t_init[idx];
    const bool peos = (y[b] == 4);
    h_sel[idx] = peos ? th : h[idx];
    if (j == 0) sent_out[b] = peos ? 1.f : sent_in[b];
}

// GRU pointwise stage 2: h_next = GRUCell(y_emb, h_sel)
__global__ void __launch_bounds__(256)
gru2(const float* __restrict__ gi, const float* __restrict__ gh,
     const float* __restrict__ h_sel, float* __restrict__ h_next)
{
    const int idx = blockIdx.x * blockDim.x + threadIdx.x;
    if (idx >= Bsz * Hsz) return;
    const int b = idx >> 10, j = idx & (Hsz - 1);
    const float ir = gi[b * G3H + j],            hr = gh[b * G3H + j];
    const float iz = gi[b * G3H + Hsz + j],      hz = gh[b * G3H + Hsz + j];
    const float in_ = gi[b * G3H + 2 * Hsz + j], hn = gh[b * G3H + 2 * Hsz + j];
    const float r = 1.f / (1.f + __expf(-(ir + hr)));
    const float z = 1.f / (1.f + __expf(-(iz + hz)));
    const float n = tanhf(in_ + r * hn);
    h_next[idx] = (1.f - z) * n + z * h_sel[idx];
}

// y_emb[b,e] = embedding[y[b], e]
__global__ void __launch_bounds__(256)
emb_gather(const int* __restrict__ y, const float* __restrict__ emb,
           float* __restrict__ y_emb)
{
    const int idx = blockIdx.x * blockDim.x + threadIdx.x;
    if (idx >= Bsz * Esz) return;
    const int b = idx >> 9, e = idx & (Esz - 1);
    y_emb[idx] = emb[(size_t)y[b] * Esz + e];
}

// scores[b,s] = mask>0 ? dot(bank[b,s,:], proj[b,:]) : -1e9
__global__ void __launch_bounds__(256)
attn_scores(const float* __restrict__ bank, const float* __restrict__ proj,
            const float* __restrict__ mask, float* __restrict__ scores)
{
    __shared__ float pj[Msz];
    const int b = blockIdx.x;
    const int s0 = blockIdx.y * 16;
    for (int i = threadIdx.x; i < Msz; i += 256) pj[i] = proj[b * Msz + i];
    __syncthreads();
    const int row = threadIdx.x >> 4;
    const int l16 = threadIdx.x & 15;
    const int s = s0 + row;
    float part = 0.f;
    if (s < Ssz) {
        const float* br = bank + ((size_t)b * Ssz + s) * Msz;
        for (int k = l16; k < Msz; k += 16) part += br[k] * pj[k];
    }
    part += __shfl_xor(part, 8);
    part += __shfl_xor(part, 4);
    part += __shfl_xor(part, 2);
    part += __shfl_xor(part, 1);
    if (s < Ssz && l16 == 0) {
        scores[b * Ssz + s] = (mask[b * Ssz + s] > 0.f) ? part : -1e9f;
    }
}

// softmax over S=400 (one block per sample)
__global__ void __launch_bounds__(512)
softmax_s(const float* __restrict__ scores, float* __restrict__ attn)
{
    __shared__ float red[512];
    const int b = blockIdx.x, t = threadIdx.x;
    const float v = (t < Ssz) ? scores[b * Ssz + t] : -3.4e38f;
    red[t] = v; __syncthreads();
    for (int off = 256; off >= 1; off >>= 1) {
        if (t < off) red[t] = fmaxf(red[t], red[t + off]);
        __syncthreads();
    }
    const float mx = red[0]; __syncthreads();
    const float e = (t < Ssz) ? __expf(v - mx) : 0.f;
    red[t] = e; __syncthreads();
    for (int off = 256; off >= 1; off >>= 1) {
        if (t < off) red[t] += red[t + off];
        __syncthreads();
    }
    const float inv = 1.f / red[0];
    if (t < Ssz) attn[b * Ssz + t] = e * inv;
}

// context[b,m] = sum_s attn[b,s]*bank[b,s,m]; also builds x=[context,h_next]
__global__ void __launch_bounds__(1024)
context_k(const float* __restrict__ bank, const float* __restrict__ attn,
          const float* __restrict__ h_next,
          float* __restrict__ ctx_out, float* __restrict__ xbuf)
{
    __shared__ float aw[Ssz];
    const int b = blockIdx.x, m = threadIdx.x;
    for (int i = threadIdx.x; i < Ssz; i += 1024) aw[i] = attn[b * Ssz + i];
    __syncthreads();
    float acc = 0.f;
    const float* bb = bank + (size_t)b * Ssz * Msz;
    for (int s = 0; s < Ssz; ++s) acc += aw[s] * bb[(size_t)s * Msz + m];
    ctx_out[b * Msz + m] = acc;
    xbuf[b * HM + m] = acc;
    xbuf[b * HM + Hsz + m] = h_next[b * Hsz + m];
}

// p_gen[b] = sigmoid(pgen_W . [ctx, h_next, y_emb] + pgen_b)
__global__ void __launch_bounds__(256)
pgen_k(const float* __restrict__ ctx, const float* __restrict__ h_next,
       const float* __restrict__ y_emb, const float* __restrict__ Wp,
       const float* __restrict__ bp, float* __restrict__ pg_out,
       float* __restrict__ pg_ws)
{
    __shared__ float red[256];
    const int b = blockIdx.x, t = threadIdx.x;
    float part = 0.f;
    for (int i = t; i < Msz; i += 256) part += Wp[i] * ctx[b * Msz + i];
    for (int i = t; i < Hsz; i += 256) part += Wp[Msz + i] * h_next[b * Hsz + i];
    for (int i = t; i < Esz; i += 256) part += Wp[Msz + Hsz + i] * y_emb[b * Esz + i];
    red[t] = part; __syncthreads();
    for (int off = 128; off >= 1; off >>= 1) {
        if (t < off) red[t] += red[t + off];
        __syncthreads();
    }
    if (t == 0) {
        const float p = 1.f / (1.f + __expf(-(red[0] + bp[0])));
        pg_out[b] = p; pg_ws[b] = p;
    }
}

// logits (stored in final_dist region) -> vocab_dist; final <- p_gen*vocab
__global__ void __launch_bounds__(1024)
vocab_softmax(float* __restrict__ logits, float* __restrict__ vocab,
              const float* __restrict__ pg)
{
    __shared__ float red[1024];
    const int b = blockIdx.x, t = threadIdx.x;
    float mx = -3.4e38f;
    for (int v = t; v < Vsz; v += 1024)
        mx = fmaxf(mx, logits[(size_t)b * Vsz + v]);
    red[t] = mx; __syncthreads();
    for (int off = 512; off >= 1; off >>= 1) {
        if (t < off) red[t] = fmaxf(red[t], red[t + off]);
        __syncthreads();
    }
    mx = red[0]; __syncthreads();
    float sum = 0.f;
    for (int v = t; v < Vsz; v += 1024)
        sum += __expf(logits[(size_t)b * Vsz + v] - mx);
    red[t] = sum; __syncthreads();
    for (int off = 512; off >= 1; off >>= 1) {
        if (t < off) red[t] += red[t + off];
        __syncthreads();
    }
    const float inv = 1.f / red[0];
    const float p = pg[b];
    for (int v = t; v < Vsz; v += 1024) {
        const float pv = __expf(logits[(size_t)b * Vsz + v] - mx) * inv;
        vocab[(size_t)b * Vsz + v] = pv;
        logits[(size_t)b * Vsz + v] = p * pv;    // becomes final_dist
    }
}

// pointer scatter: final[b, src[b,s]] += (1-pg[b]) * attn[b,s]
__global__ void __launch_bounds__(256)
scatter_k(const int* __restrict__ src, const float* __restrict__ attn,
          const float* __restrict__ pg, float* __restrict__ fin)
{
    const int idx = blockIdx.x * blockDim.x + threadIdx.x;
    if (idx >= Bsz * Ssz) return;
    const int b = idx / Ssz;
    atomicAdd(&fin[(size_t)b * Vsz + src[idx]], (1.f - pg[b]) * attn[idx]);
}

__global__ void __launch_bounds__(256)
copy_k(const float* __restrict__ in, float* __restrict__ out, int n)
{
    const int idx = blockIdx.x * blockDim.x + threadIdx.x;
    if (idx < n) out[idx] = in[idx];
}

// ---------------------------------------------------------------------------
extern "C" void kernel_launch(void* const* d_in, const int* in_sizes, int n_in,
                              void* d_out, int out_size, void* d_ws, size_t ws_size,
                              hipStream_t stream)
{
    (void)in_sizes; (void)n_in; (void)out_size; (void)ws_size;

    const int*   y        = (const int*)  d_in[0];
    const float* h        = (const float*)d_in[1];
    const float* bank     = (const float*)d_in[2];
    const float* src_mask = (const float*)d_in[3];
    const int*   src      = (const int*)  d_in[4];
    const int*   aspect   = (const int*)  d_in[5];
    const float* h_t_init = (const float*)d_in[7];
    const float* sent_in  = (const float*)d_in[8];
    const float* coverage = (const float*)d_in[9];
    const float* wasp     = (const float*)d_in[10];
    const float* emb      = (const float*)d_in[11];
    const float* gru_W_ih = (const float*)d_in[12];
    const float* gru_W_hh = (const float*)d_in[13];
    const float* gru_b_ih = (const float*)d_in[14];
    const float* gru_b_hh = (const float*)d_in[15];
    const float* g_W_ih   = (const float*)d_in[16];
    const float* g_W_hh   = (const float*)d_in[17];
    const float* g_b_ih   = (const float*)d_in[18];
    const float* g_b_hh   = (const float*)d_in[19];
    const float* attn_W   = (const float*)d_in[20];
    const float* attn_b   = (const float*)d_in[21];
    const float* pgen_W   = (const float*)d_in[22];
    const float* pgen_b   = (const float*)d_in[23];
    const float* W1       = (const float*)d_in[24];
    const float* b1       = (const float*)d_in[25];
    const float* W2       = (const float*)d_in[26];
    const float* b2       = (const float*)d_in[27];

    // ---- output layout (flat f32, reference tuple order) ----
    float* out       = (float*)d_out;
    float* out_vocab = out;                                   // [B,V]
    float* out_final = out + (size_t)Bsz * Vsz;               // [B,V]
    float* out_h     = out_final + (size_t)Bsz * Vsz;         // [B,H]
    float* out_ctx   = out_h   + Bsz * Hsz;                   // [B,M]
    float* out_attn  = out_ctx + Bsz * Msz;                   // [B,S]
    float* out_pg    = out_attn + Bsz * Ssz;                  // [B,1]
    float* out_cov   = out_pg + Bsz;                          // [B,S]
    float* out_sent  = out_cov + Bsz * Ssz;                   // [B]
    float* out_wasp  = out_sent + Bsz;                        // [B,M]

    // ---- workspace layout ----
    float* ws     = (float*)d_ws;
    float* gi     = ws;                         // [B,3H]
    float* gh     = gi  + Bsz * G3H;            // [B,3H]
    float* h_sel  = gh  + Bsz * G3H;            // [B,H]
    float* y_emb  = h_sel + Bsz * Hsz;          // [B,E]
    float* proj   = y_emb + Bsz * Esz;          // [B,M]
    float* scores = proj  + Bsz * Msz;          // [B,S]
    float* xbuf   = scores + Bsz * Ssz;         // [B,H+M]
    float* h1     = xbuf + Bsz * HM;            // [B,H]
    float* pg_ws  = h1   + Bsz * Hsz;           // [B]
    int*   counts = (int*)(pg_ws + Bsz);        // [A]
    int*   lists  = counts + Asz;               // [A,B]
    int*   idlist = lists + Asz * Bsz;          // [B]

    const dim3 blk128(128);

    // 0) aspect routing table + identity list
    aspect_build<<<1, 64, 0, stream>>>(aspect, counts, lists, idlist);

    // 1) global GRU gates
    gemm_wmma_f16<<<dim3(G3H / 128, 4, 1), blk128, 0, stream>>>(
        h, Hsz, g_W_ih, g_b_ih, gi, G3H, Bsz, G3H, Hsz, idlist, nullptr, 0, 0);
    gemm_wmma_f16<<<dim3(G3H / 128, 4, 1), blk128, 0, stream>>>(
        h_t_init, Hsz, g_W_hh, g_b_hh, gh, G3H, Bsz, G3H, Hsz, idlist, nullptr, 0, 0);
    gru1_select<<<(Bsz * Hsz + 255) / 256, 256, 0, stream>>>(
        gi, gh, h, h_t_init, y, sent_in, h_sel, out_sent);

    // 2) embedding lookup + main GRU
    emb_gather<<<(Bsz * Esz + 255) / 256, 256, 0, stream>>>(y, emb, y_emb);
    gemm_wmma_f16<<<dim3(G3H / 128, 4, 1), blk128, 0, stream>>>(
        y_emb, Esz, gru_W_ih, gru_b_ih, gi, G3H, Bsz, G3H, Esz, idlist, nullptr, 0, 0);
    gemm_wmma_f16<<<dim3(G3H / 128, 4, 1), blk128, 0, stream>>>(
        h_sel, Hsz, gru_W_hh, gru_b_hh, gh, G3H, Bsz, G3H, Hsz, idlist, nullptr, 0, 0);
    gru2<<<(Bsz * Hsz + 255) / 256, 256, 0, stream>>>(gi, gh, h_sel, out_h);

    // 3) attention
    gemm_wmma_f16<<<dim3(Msz / 128, 4, 1), blk128, 0, stream>>>(
        out_h, Hsz, attn_W, attn_b, proj, Msz, Bsz, Msz, Hsz, idlist, nullptr, 0, 0);
    attn_scores<<<dim3(Bsz, (Ssz + 15) / 16), 256, 0, stream>>>(
        bank, proj, src_mask, scores);
    softmax_s<<<Bsz, 512, 0, stream>>>(scores, out_attn);
    context_k<<<Bsz, 1024, 0, stream>>>(bank, out_attn, out_h, out_ctx, xbuf);

    // 4) copy gate
    pgen_k<<<Bsz, 256, 0, stream>>>(out_ctx, out_h, y_emb, pgen_W, pgen_b,
                                    out_pg, pg_ws);

    // 5) aspect-routed vocab head (each expert's weights read once)
    gemm_wmma_f16<<<dim3(Hsz / 128, 4, Asz), blk128, 0, stream>>>(
        xbuf, HM, W1, b1, h1, Hsz, 0, Hsz, HM, lists, counts,
        (long long)Hsz * HM, (long long)Hsz);
    gemm_wmma_f16<<<dim3(Vsz / 128, 4, Asz), blk128, 0, stream>>>(
        h1, Hsz, W2, b2, out_final /*logits staging*/, Vsz, 0, Vsz, Hsz,
        lists, counts, (long long)Vsz * Hsz, (long long)Vsz);

    // 6) vocab softmax + pointer mixture
    vocab_softmax<<<Bsz, 1024, 0, stream>>>(out_final, out_vocab, pg_ws);
    scatter_k<<<(Bsz * Ssz + 255) / 256, 256, 0, stream>>>(src, out_attn,
                                                           pg_ws, out_final);

    // 7) pass-through outputs
    copy_k<<<(Bsz * Ssz + 255) / 256, 256, 0, stream>>>(coverage, out_cov,
                                                        Bsz * Ssz);
    copy_k<<<(Bsz * Msz + 255) / 256, 256, 0, stream>>>(wasp, out_wasp,
                                                        Bsz * Msz);
}
